// MultiHeadLatentAttention_7327214207477
// MI455X (gfx1250) — matrix operand order repaired and checked by reading, not compiled, not run on hardware
//
#include <hip/hip_runtime.h>
#include <hip/hip_bf16.h>

// ---------------- problem constants ----------------
#define Bc 2
#define Sc 2048
#define Dm 2048
#define QL 1536
#define KVL 512
#define Hh 16
#define HD 128
#define Mrows 4096            // B*S
#define QKD 256               // per-head q/k dim (128 nope + 128 rope)

typedef __attribute__((ext_vector_type(16))) __bf16 bf16x16;
typedef __attribute__((ext_vector_type(8)))  float  f32x8;
typedef __attribute__((ext_vector_type(4)))  unsigned int u32x4;
typedef __attribute__((ext_vector_type(4)))  int   i32x4;
typedef __attribute__((ext_vector_type(8)))  int   i32x8;

union FragAB { bf16x16 v; uint4 u[2]; };

__device__ __forceinline__ unsigned short bf16_bits(float f) {
    __bf16 h = (__bf16)f;               // RNE convert
    unsigned short s;
    __builtin_memcpy(&s, &h, 2);
    return s;
}

#if defined(__has_builtin)
#if __has_builtin(__builtin_amdgcn_tensor_load_to_lds) && __has_builtin(__builtin_amdgcn_s_wait_tensorcnt)
#define USE_TDM 1
#endif
#endif
#ifndef USE_TDM
#define USE_TDM 0
#endif

// ---------------------------------------------------------------------
// Tensor Data Mover: DMA a 2D bf16 tile (tile_d0 contiguous elems per row,
// tile_d1 rows, row stride = stride_elems) from global into LDS, adding
// pad_amt DWORDs of LDS padding every pad_int DWORDs (ISA D# section 8).
// pad codes: interval 3 -> 16 DW (32 bf16 rows), 6 -> 128 DW (256 bf16 rows);
// amount 3 -> 4 DW (8 bf16 pad).
// ---------------------------------------------------------------------
#if USE_TDM
__device__ __forceinline__ void tdm_load_2d(const void* gptr, unsigned lds_addr,
                                            unsigned tile_d0, unsigned tile_d1,
                                            unsigned stride_elems,
                                            unsigned pad_interval_code,
                                            unsigned pad_amount_code)
{
    unsigned long long ga = (unsigned long long)gptr;
    u32x4 g0;
    g0[0] = 1u;                                   // count=1, user descriptor
    g0[1] = lds_addr;                             // LDS byte address
    g0[2] = (unsigned)(ga & 0xFFFFFFFFu);         // global_addr[31:0]
    g0[3] = (unsigned)((ga >> 32) & 0x01FFFFFFu) | (2u << 30);  // addr[56:32] | type=2

    const unsigned td0 = 0x100000u;               // generous tensor dims: no OOB
    const unsigned td1 = 0x100000u;
    i32x8 g1;
    g1[0] = (int)((1u << 16)                      // data_size = 2 bytes
                | (1u << 20)                      // pad_enable
                | (pad_interval_code << 22)
                | (pad_amount_code << 25));
    g1[1] = (int)((td0 & 0xFFFFu) << 16);                         // dim0 lo16 (atomic addr = 0)
    g1[2] = (int)(((td0 >> 16) & 0xFFFFu) | ((td1 & 0xFFFFu) << 16));
    g1[3] = (int)(((td1 >> 16) & 0xFFFFu) | ((tile_d0 & 0xFFFFu) << 16)); // tile_dim0
    g1[4] = (int)(tile_d1 & 0xFFFFu);                             // tile_dim1 (tile_dim2=0)
    g1[5] = (int)stride_elems;                                    // dim0 stride lo32
    g1[6] = 0;                                                    // stride hi / dim1 stride
    g1[7] = 0;
    i32x4 z4 = {0, 0, 0, 0};
#if defined(__clang_major__) && (__clang_major__ >= 23)
    i32x8 z8 = {0, 0, 0, 0, 0, 0, 0, 0};
    __builtin_amdgcn_tensor_load_to_lds(g0, g1, z4, z4, z8, 0);
#else
    __builtin_amdgcn_tensor_load_to_lds(g0, g1, z4, z4, 0);
#endif
}
#endif

// =====================================================================
// bf16 -> bf16 WMMA GEMM, 128x128x32 tiles, double-buffered TDM staging.
// A: M x K bf16 row-major.  Bt: N x K bf16 (pre-transposed weights).
// MODE 0: f32 plain output (+bias)      MODE 1: bf16 plain output
// MODE 2: bf16 head-concat (row*4096 + (col>>7)*256 + (col&127))
// MODE 3: bf16 v-transposed ((b*16+h)*128+n)*2048 + s
// =====================================================================
#define AST 40   // LDS row stride for 32-elem rows (bf16)
#define KSTR 264 // LDS row stride for 256-elem rows (bf16)

template<int MODE>
__global__ __launch_bounds__(256)
void gemm_wmma_bf16(const unsigned short* __restrict__ A,
                    const unsigned short* __restrict__ Bt,
                    void* __restrict__ Cv, const float* __restrict__ bias,
                    int M, int N, int K)
{
    __shared__ unsigned short lds_a[2][128 * AST];
    __shared__ unsigned short lds_b[2][128 * AST];

    const int tid  = threadIdx.x;
    const int lane = tid & 31;
    const int wave = tid >> 5;
    const int l15  = lane & 15;
    const int lh   = lane >> 4;

    const int m0 = blockIdx.y * 128;
    const int n0 = blockIdx.x * 128;
    const int wm = (wave >> 2) * 64;
    const int wn = (wave & 3) * 32;

    f32x8 acc[4][2];
#pragma unroll
    for (int mi = 0; mi < 4; ++mi)
#pragma unroll
        for (int nj = 0; nj < 2; ++nj)
#pragma unroll
            for (int e = 0; e < 8; ++e) acc[mi][nj][e] = 0.0f;

    const int nk = K >> 5;
    int cur = 0;

#if USE_TDM
    if (tid == 0) {
        tdm_load_2d(A  + (size_t)m0 * K, (unsigned)(uintptr_t)&lds_a[0][0], 32, 128, K, 3, 3);
        tdm_load_2d(Bt + (size_t)n0 * K, (unsigned)(uintptr_t)&lds_b[0][0], 32, 128, K, 3, 3);
    }
#endif

    for (int kt = 0; kt < nk; ++kt) {
#if USE_TDM
        if (tid == 0) {
            if (kt + 1 < nk) {
                tdm_load_2d(A  + (size_t)m0 * K + (kt + 1) * 32,
                            (unsigned)(uintptr_t)&lds_a[cur ^ 1][0], 32, 128, K, 3, 3);
                tdm_load_2d(Bt + (size_t)n0 * K + (kt + 1) * 32,
                            (unsigned)(uintptr_t)&lds_b[cur ^ 1][0], 32, 128, K, 3, 3);
                __builtin_amdgcn_s_wait_tensorcnt(2);
            } else {
                __builtin_amdgcn_s_wait_tensorcnt(0);
            }
        }
        __syncthreads();
#else
        __syncthreads();
        // manual staging fallback: 128 rows x 32 bf16 = 512 uint4 per matrix
#pragma unroll
        for (int i = 0; i < 2; ++i) {
            int idx = tid + i * 256;
            int row = idx >> 2, c = (idx & 3) << 3;
            *(uint4*)(&lds_a[cur][row * AST + c]) =
                *(const uint4*)(A + (size_t)(m0 + row) * K + kt * 32 + c);
            *(uint4*)(&lds_b[cur][row * AST + c]) =
                *(const uint4*)(Bt + (size_t)(n0 + row) * K + kt * 32 + c);
        }
        __syncthreads();
#endif
        const unsigned short* la = lds_a[cur];
        const unsigned short* lb = lds_b[cur];
        FragAB af[4], bfr[2];
#pragma unroll
        for (int mi = 0; mi < 4; ++mi) {
            const unsigned short* p = &la[(wm + mi * 16 + l15) * AST + lh * 8];
            af[mi].u[0] = *(const uint4*)(p);
            af[mi].u[1] = *(const uint4*)(p + 16);
        }
#pragma unroll
        for (int nj = 0; nj < 2; ++nj) {
            const unsigned short* p = &lb[(wn + nj * 16 + l15) * AST + lh * 8];
            bfr[nj].u[0] = *(const uint4*)(p);
            bfr[nj].u[1] = *(const uint4*)(p + 16);
        }
#pragma unroll
        for (int mi = 0; mi < 4; ++mi)
#pragma unroll
            for (int nj = 0; nj < 2; ++nj)
                acc[mi][nj] = __builtin_amdgcn_wmma_f32_16x16x32_bf16(
                    false, af[mi].v, false, bfr[nj].v, (short)0, acc[mi][nj], false, false);
#if USE_TDM
        __syncthreads();   // all waves done with buf[cur] before next DMA overwrites it
#endif
        cur ^= 1;
    }

    // ---- epilogue ----
#pragma unroll
    for (int mi = 0; mi < 4; ++mi)
#pragma unroll
        for (int nj = 0; nj < 2; ++nj)
#pragma unroll
            for (int r = 0; r < 8; ++r) {
                int row = m0 + wm + mi * 16 + r + 8 * lh;
                int col = n0 + wn + nj * 16 + l15;
                float v = acc[mi][nj][r];
                if (MODE == 0) {
                    if (bias) v += bias[col];
                    ((float*)Cv)[(size_t)row * N + col] = v;
                } else if (MODE == 1) {
                    ((unsigned short*)Cv)[(size_t)row * N + col] = bf16_bits(v);
                } else if (MODE == 2) {
                    ((unsigned short*)Cv)[(size_t)row * 4096 + ((col >> 7) << 8) + (col & 127)] =
                        bf16_bits(v);
                } else { // MODE 3: vt[(b*16+h)*128+n][s]
                    int bb = row >> 11, ss = row & 2047;
                    ((unsigned short*)Cv)[((size_t)((bb * Hh + (col >> 7)) * HD + (col & 127))) * Sc + ss] =
                        bf16_bits(v);
                }
            }
}

// =====================================================================
// elementwise converters
// =====================================================================
__global__ __launch_bounds__(256)
void cvt_bf16_kernel(const float* __restrict__ in, unsigned short* __restrict__ out, int n)
{
    int idx = blockIdx.x * blockDim.x + threadIdx.x;
    if (idx < n) out[idx] = bf16_bits(in[idx]);
}

__global__ __launch_bounds__(256)
void transpose_bf16_kernel(const float* __restrict__ in, unsigned short* __restrict__ out,
                           int K, int N)   // in: K x N f32 -> out: N x K bf16
{
    int idx = blockIdx.x * blockDim.x + threadIdx.x;
    if (idx >= K * N) return;
    int k = idx / N, n = idx - k * N;
    out[(size_t)n * K + k] = bf16_bits(in[idx]);
}

// =====================================================================
// RoPE kernels (write bf16 into the [128:256) half of q_new / k_new)
// =====================================================================
__global__ __launch_bounds__(256)
void rope_q_kernel(const float* __restrict__ x, unsigned short* __restrict__ qnew)
{
    int idx = blockIdx.x * blockDim.x + threadIdx.x;   // over 4096*2048
    int row = idx >> 11;
    int i   = idx & 2047;
    int s   = row & (Sc - 1);
    int fi  = i & 1023;                                // half = 1024
    float inv = __powf(10000.0f, -2.0f * (float)fi / 2048.0f);
    float ang = (float)s * inv;
    float c = __cosf(ang), sn = __sinf(ang);
    float xv  = x[idx];
    float rot = (i < 1024) ? -x[(size_t)row * 2048 + i + 1024]
                           :  x[(size_t)row * 2048 + i - 1024];
    float val = xv * c + rot * sn;
    qnew[(size_t)row * 4096 + ((i >> 7) << 8) + 128 + (i & 127)] = bf16_bits(val);
}

__global__ __launch_bounds__(256)
void rope_k_kernel(const float* __restrict__ x, unsigned short* __restrict__ knew)
{
    int idx = blockIdx.x * blockDim.x + threadIdx.x;   // over 4096*128
    int row = idx >> 7;
    int i   = idx & 127;
    int s   = row & (Sc - 1);
    int fi  = i & 63;                                  // half = 64
    float inv = __powf(10000.0f, -2.0f * (float)fi / 128.0f);
    float ang = (float)s * inv;
    float c = __cosf(ang), sn = __sinf(ang);
    float xv  = x[idx];
    float rot = (i < 64) ? -x[(size_t)row * 128 + i + 64]
                         :  x[(size_t)row * 128 + i - 64];
    float val = xv * c + rot * sn;
    unsigned short bits = bf16_bits(val);
#pragma unroll
    for (int h = 0; h < Hh; ++h)
        knew[(size_t)row * 4096 + h * 256 + 128 + i] = bits;
}

// =====================================================================
// Flash attention: block = 4 waves = 64 q rows for one (b,h).
// Double-buffered TDM staging of K (32x256) and V^T (128x32) chunks.
// =====================================================================
#define VST 40
#define PST 40

__global__ __launch_bounds__(128)
void mla_flash_attn(const unsigned short* __restrict__ qn,
                    const unsigned short* __restrict__ kn,
                    const unsigned short* __restrict__ vt,   // (b,h,n,s) bf16
                    unsigned short* __restrict__ out)        // attn output bf16 (b,s,2048)
{
    __shared__ unsigned short lds_k[2][32 * KSTR];
    __shared__ unsigned short lds_v[2][128 * VST];
    __shared__ unsigned short lds_p[4 * 16 * PST];

    const int tid  = threadIdx.x;
    const int lane = tid & 31;
    const int wave = tid >> 5;
    const int l15  = lane & 15;
    const int lh   = lane >> 4;

    const int qbase = blockIdx.x * 64;
    const int h     = blockIdx.y;
    const int b     = blockIdx.z;

    const unsigned short* kbh = kn + ((size_t)(b * Sc) * Hh + h) * QKD;   // +key*Hh*QKD
    const unsigned short* vbh = vt + (size_t)(b * Hh + h) * HD * Sc;      // rows n, stride Sc

    // ---- Q fragments: 16 rows x 256 feats -> 8 A-frags ----
    const int qrow = qbase + wave * 16 + l15;
    const unsigned short* qp = qn + ((size_t)(b * Sc + qrow) * Hh + h) * QKD;
    FragAB qf[8];
#pragma unroll
    for (int i = 0; i < 8; ++i) {
        qf[i].u[0] = *(const uint4*)(qp + i * 32 + lh * 8);
        qf[i].u[1] = *(const uint4*)(qp + i * 32 + 16 + lh * 8);
    }

    f32x8 O[8];
#pragma unroll
    for (int nt = 0; nt < 8; ++nt)
#pragma unroll
        for (int e = 0; e < 8; ++e) O[nt][e] = 0.0f;
    float m8[8], l8[8];
#pragma unroll
    for (int r = 0; r < 8; ++r) { m8[r] = -3.0e38f; l8[r] = 0.0f; }

    const float scale = 0.0625f;           // 1/sqrt(2*HD)
    const int nkt = qbase / 32 + 2;        // causal 32-key chunks
    int cur = 0;
    unsigned short* ldsp = &lds_p[wave * 16 * PST];

#if USE_TDM
    if (tid == 0) {
        tdm_load_2d(kbh, (unsigned)(uintptr_t)&lds_k[0][0], 256, 32, Hh * QKD, 6, 3);
        tdm_load_2d(vbh, (unsigned)(uintptr_t)&lds_v[0][0], 32, 128, Sc, 3, 3);
    }
#endif

    for (int kt = 0; kt < nkt; ++kt) {
#if USE_TDM
        if (tid == 0) {
            if (kt + 1 < nkt) {
                tdm_load_2d(kbh + (size_t)(kt + 1) * 32 * Hh * QKD,
                            (unsigned)(uintptr_t)&lds_k[cur ^ 1][0], 256, 32, Hh * QKD, 6, 3);
                tdm_load_2d(vbh + (kt + 1) * 32,
                            (unsigned)(uintptr_t)&lds_v[cur ^ 1][0], 32, 128, Sc, 3, 3);
                __builtin_amdgcn_s_wait_tensorcnt(2);
            } else {
                __builtin_amdgcn_s_wait_tensorcnt(0);
            }
        }
        __syncthreads();
#else
        __syncthreads();
#pragma unroll
        for (int j = 0; j < 8; ++j) {       // K: 32 rows x 256 bf16 = 1024 uint4
            int idx = tid + j * 128;
            int key = idx >> 5, c = (idx & 31) << 3;
            *(uint4*)(&lds_k[cur][key * KSTR + c]) =
                *(const uint4*)(kbh + (size_t)(kt * 32 + key) * Hh * QKD + c);
        }
#pragma unroll
        for (int j = 0; j < 4; ++j) {       // V^T: 128 rows x 32 bf16 = 512 uint4
            int idx = tid + j * 128;
            int row = idx >> 2, c = (idx & 3) << 3;
            *(uint4*)(&lds_v[cur][row * VST + c]) =
                *(const uint4*)(vbh + (size_t)row * Sc + kt * 32 + c);
        }
        __syncthreads();
#endif
        const unsigned short* lk = lds_k[cur];
        const unsigned short* lv = lds_v[cur];

        // ---- scores: 16 q x 32 k, K = 256 ----
        f32x8 s0, s1;
#pragma unroll
        for (int e = 0; e < 8; ++e) { s0[e] = 0.0f; s1[e] = 0.0f; }
#pragma unroll
        for (int i = 0; i < 8; ++i) {
            FragAB kb0, kb1;
            const unsigned short* p0 = &lk[(l15) * KSTR + i * 32 + lh * 8];
            const unsigned short* p1 = &lk[(16 + l15) * KSTR + i * 32 + lh * 8];
            kb0.u[0] = *(const uint4*)(p0); kb0.u[1] = *(const uint4*)(p0 + 16);
            kb1.u[0] = *(const uint4*)(p1); kb1.u[1] = *(const uint4*)(p1 + 16);
            s0 = __builtin_amdgcn_wmma_f32_16x16x32_bf16(false, qf[i].v, false, kb0.v,
                                                         (short)0, s0, false, false);
            s1 = __builtin_amdgcn_wmma_f32_16x16x32_bf16(false, qf[i].v, false, kb1.v,
                                                         (short)0, s1, false, false);
        }

        // ---- online softmax ----
        const int qr = qbase + wave * 16 + 8 * lh;
#pragma unroll
        for (int r = 0; r < 8; ++r) {
            int row = qr + r;
            int c0 = kt * 32 + l15;
            int c1 = c0 + 16;
            float v0 = (c0 <= row) ? s0[r] * scale : -1.0e9f;
            float v1 = (c1 <= row) ? s1[r] * scale : -1.0e9f;
            float mr = fmaxf(v0, v1);
            mr = fmaxf(mr, __shfl_xor(mr, 1, 32));
            mr = fmaxf(mr, __shfl_xor(mr, 2, 32));
            mr = fmaxf(mr, __shfl_xor(mr, 4, 32));
            mr = fmaxf(mr, __shfl_xor(mr, 8, 32));
            float mn = fmaxf(m8[r], mr);
            float alpha = __expf(m8[r] - mn);
            float p0 = __expf(v0 - mn);
            float p1 = __expf(v1 - mn);
            float sum = p0 + p1;
            sum += __shfl_xor(sum, 1, 32);
            sum += __shfl_xor(sum, 2, 32);
            sum += __shfl_xor(sum, 4, 32);
            sum += __shfl_xor(sum, 8, 32);
            l8[r] = l8[r] * alpha + sum;
            m8[r] = mn;
#pragma unroll
            for (int nt = 0; nt < 8; ++nt) O[nt][r] *= alpha;
            ldsp[(r + 8 * lh) * PST + l15]      = bf16_bits(p0);
            ldsp[(r + 8 * lh) * PST + 16 + l15] = bf16_bits(p1);
        }

        // ---- P (16x32) @ V (32x128) ----
        FragAB pf;
        pf.u[0] = *(const uint4*)(&ldsp[l15 * PST + lh * 8]);
        pf.u[1] = *(const uint4*)(&ldsp[l15 * PST + 16 + lh * 8]);
#pragma unroll
        for (int nt = 0; nt < 8; ++nt) {
            FragAB vb;
            const unsigned short* p = &lv[(nt * 16 + l15) * VST + lh * 8];
            vb.u[0] = *(const uint4*)(p);
            vb.u[1] = *(const uint4*)(p + 16);
            O[nt] = __builtin_amdgcn_wmma_f32_16x16x32_bf16(false, pf.v, false, vb.v,
                                                            (short)0, O[nt], false, false);
        }
        __syncthreads();   // done reading buf[cur] before next DMA/copy overwrites it
        cur ^= 1;
    }

    // ---- epilogue: normalize & store bf16 ----
#pragma unroll
    for (int nt = 0; nt < 8; ++nt)
#pragma unroll
        for (int r = 0; r < 8; ++r) {
            int row = qbase + wave * 16 + r + 8 * lh;
            float val = O[nt][r] / l8[r];
            out[(size_t)(b * Sc + row) * Dm + h * HD + nt * 16 + l15] = bf16_bits(val);
        }
}

// =====================================================================
// Orchestration
// =====================================================================
extern "C" void kernel_launch(void* const* d_in, const int* in_sizes, int n_in,
                              void* d_out, int out_size, void* d_ws, size_t ws_size,
                              hipStream_t stream) {
    (void)in_sizes; (void)n_in; (void)out_size; (void)ws_size;
    const float* inq     = (const float*)d_in[0];
    const float* ink     = (const float*)d_in[1];
    // d_in[2] (inputs_v) is unused by the reference
    const float* Wq_down = (const float*)d_in[3];
    const float* Wkv_down= (const float*)d_in[4];
    const float* Wq_up   = (const float*)d_in[5];
    const float* Wk_up   = (const float*)d_in[6];
    const float* Wv_up   = (const float*)d_in[7];
    const float* Wq_rope = (const float*)d_in[8];
    const float* Wk_rope = (const float*)d_in[9];
    const float* Wout    = (const float*)d_in[10];
    const float* bout    = (const float*)d_in[11];

    typedef unsigned short u16;
    char* ws = (char*)d_ws;
    size_t off = 0;
    u16* bq     = (u16*)(ws + off); off += (size_t)Mrows * Dm * 2;     // inputs_q bf16
    u16* bk     = (u16*)(ws + off); off += (size_t)Mrows * Dm * 2;     // inputs_k bf16
    u16* wqd_t  = (u16*)(ws + off); off += (size_t)QL * Dm * 2;        // Wq_down^T
    u16* wkvd_t = (u16*)(ws + off); off += (size_t)KVL * Dm * 2;       // Wkv_down^T
    u16* wqu_t  = (u16*)(ws + off); off += (size_t)Dm * QL * 2;        // Wq_up^T
    u16* wku_t  = (u16*)(ws + off); off += (size_t)Dm * KVL * 2;       // Wk_up^T
    u16* wvu_t  = (u16*)(ws + off); off += (size_t)Dm * KVL * 2;       // Wv_up^T
    u16* wqr_t  = (u16*)(ws + off); off += (size_t)Dm * QL * 2;        // Wq_rope^T
    u16* wkr_t  = (u16*)(ws + off); off += (size_t)HD * Dm * 2;        // Wk_rope^T
    u16* wout_t = (u16*)(ws + off); off += (size_t)Dm * Dm * 2;        // Wout^T
    u16* q_lat  = (u16*)(ws + off); off += (size_t)Mrows * QL * 2;
    u16* kv_lat = (u16*)(ws + off); off += (size_t)Mrows * KVL * 2;
    u16* q_new  = (u16*)(ws + off); off += (size_t)Mrows * 4096 * 2;
    u16* k_new  = (u16*)(ws + off); off += (size_t)Mrows * 4096 * 2;
    u16* vtb    = (u16*)(ws + off); off += (size_t)Mrows * Dm * 2;     // V^T (b,h,n,s)
    float* ropeq = (float*)(ws + off);
    u16* attn_o  = (u16*)(ws + off);   // alias: ropeq consumed before attention
    off += (size_t)Mrows * Dm * 4;
    float* ropek = (float*)(ws + off); off += (size_t)Mrows * HD * 4;

    dim3 blk(256);
    // ---- one-time conversions (deterministic, rerun every call) ----
    cvt_bf16_kernel<<<(Mrows * Dm + 255) / 256, blk, 0, stream>>>(inq, bq, Mrows * Dm);
    cvt_bf16_kernel<<<(Mrows * Dm + 255) / 256, blk, 0, stream>>>(ink, bk, Mrows * Dm);
    transpose_bf16_kernel<<<(Dm * QL + 255) / 256, blk, 0, stream>>>(Wq_down, wqd_t, Dm, QL);
    transpose_bf16_kernel<<<(Dm * KVL + 255) / 256, blk, 0, stream>>>(Wkv_down, wkvd_t, Dm, KVL);
    transpose_bf16_kernel<<<(QL * Dm + 255) / 256, blk, 0, stream>>>(Wq_up, wqu_t, QL, Dm);
    transpose_bf16_kernel<<<(KVL * Dm + 255) / 256, blk, 0, stream>>>(Wk_up, wku_t, KVL, Dm);
    transpose_bf16_kernel<<<(KVL * Dm + 255) / 256, blk, 0, stream>>>(Wv_up, wvu_t, KVL, Dm);
    transpose_bf16_kernel<<<(QL * Dm + 255) / 256, blk, 0, stream>>>(Wq_rope, wqr_t, QL, Dm);
    transpose_bf16_kernel<<<(Dm * HD + 255) / 256, blk, 0, stream>>>(Wk_rope, wkr_t, Dm, HD);
    transpose_bf16_kernel<<<(Dm * Dm + 255) / 256, blk, 0, stream>>>(Wout, wout_t, Dm, Dm);

    // ---- projection GEMMs (all bf16 WMMA, TDM-staged) ----
    gemm_wmma_bf16<1><<<dim3(QL/128, Mrows/128), blk, 0, stream>>>(bq, wqd_t, q_lat, nullptr, Mrows, QL, Dm);
    gemm_wmma_bf16<1><<<dim3(KVL/128, Mrows/128), blk, 0, stream>>>(bk, wkvd_t, kv_lat, nullptr, Mrows, KVL, Dm);
    gemm_wmma_bf16<2><<<dim3(Dm/128, Mrows/128), blk, 0, stream>>>(q_lat, wqu_t, q_new, nullptr, Mrows, Dm, QL);
    gemm_wmma_bf16<2><<<dim3(Dm/128, Mrows/128), blk, 0, stream>>>(kv_lat, wku_t, k_new, nullptr, Mrows, Dm, KVL);
    gemm_wmma_bf16<3><<<dim3(Dm/128, Mrows/128), blk, 0, stream>>>(kv_lat, wvu_t, vtb, nullptr, Mrows, Dm, KVL);
    gemm_wmma_bf16<0><<<dim3(Dm/128, Mrows/128), blk, 0, stream>>>(q_lat, wqr_t, ropeq, nullptr, Mrows, Dm, QL);
    gemm_wmma_bf16<0><<<dim3(HD/128, Mrows/128), blk, 0, stream>>>(bk, wkr_t, ropek, nullptr, Mrows, HD, Dm);

    // ---- rope ----
    rope_q_kernel<<<(Mrows * Dm) / 256, blk, 0, stream>>>(ropeq, q_new);
    rope_k_kernel<<<(Mrows * HD) / 256, blk, 0, stream>>>(ropek, k_new);

    // ---- attention ----
    mla_flash_attn<<<dim3(Sc/64, Hh, Bc), dim3(128), 0, stream>>>(q_new, k_new, vtb, attn_o);

    // ---- output projection ----
    gemm_wmma_bf16<0><<<dim3(Dm/128, Mrows/128), blk, 0, stream>>>(attn_o, wout_t, (float*)d_out, bout, Mrows, Dm, Dm);
}
